// KANLayer_55551107006694
// MI455X (gfx1250) — compile-verified, matrix operand order checked by reading, not run on hardware
//
#include <hip/hip_runtime.h>
#include <hip/hip_bf16.h>
#include <math.h>

// ---------------------------------------------------------------------------
// KAN layer fused as one f16 WMMA GEMM with K = 1024 (base) + 8192 (spline).
// GEMM stages tiles through LDS with the CDNA5 Tensor Data Mover
// (tensor_load_to_lds + TENSORcnt), double-buffered; TDM pad_enable inserts
// the 16B/row anti-bank-conflict padding in hardware. WMMA operands are read
// from LDS; 64x64 wave tiles -> 16 v_wmma per 16 ds_load_b128.
// ---------------------------------------------------------------------------

typedef __attribute__((ext_vector_type(16))) _Float16 v16h;
typedef __attribute__((ext_vector_type(8)))  _Float16 v8h;
typedef __attribute__((ext_vector_type(8)))  float    v8f;
typedef __attribute__((ext_vector_type(4)))  unsigned int v4u;
typedef __attribute__((ext_vector_type(8)))  int      v8i;
typedef __attribute__((ext_vector_type(4)))  int      v4i;

#define BATCH        4096
#define IN_F         1024
#define OUT_F        1024
#define NCOEF        8          // GRID_SIZE + SPLINE_ORDER
#define NKNOT        12         // GRID_SIZE + 2*SPLINE_ORDER + 1
#define K_TOTAL      9216       // IN_F + IN_F*NCOEF
#define A_BYTES      ((size_t)BATCH * K_TOTAL * 2)   // 75,497,472

#define BM   128
#define BN   128
#define BK   64
#define LDK  72   // LDS row stride in halves: 128B data + 16B TDM pad = 144B

// ------------------------- activation / basis packing ----------------------
__global__ __launch_bounds__(256) void kan_pack_act(
    const float* __restrict__ x, const float* __restrict__ grid,
    _Float16* __restrict__ Apack)
{
    int idx = blockIdx.x * 256 + threadIdx.x;        // one thread per (b, i)
    int b = idx >> 10;
    int i = idx & (IN_F - 1);
    if (b >= BATCH) return;

    float xv = x[(size_t)b * IN_F + i];

    float t[NKNOT];
#pragma unroll
    for (int j = 0; j < NKNOT; ++j) t[j] = grid[(size_t)i * NKNOT + j];

    // Cox-de Boor up to cubic, matching the reference recursion exactly
    float b0[11], b1[10], b2[9], b3[8];
#pragma unroll
    for (int j = 0; j < 11; ++j) b0[j] = (xv >= t[j] && xv < t[j + 1]) ? 1.f : 0.f;
#pragma unroll
    for (int j = 0; j < 10; ++j)
        b1[j] = (xv - t[j]) / (t[j + 1] - t[j]) * b0[j]
              + (t[j + 2] - xv) / (t[j + 2] - t[j + 1]) * b0[j + 1];
#pragma unroll
    for (int j = 0; j < 9; ++j)
        b2[j] = (xv - t[j]) / (t[j + 2] - t[j]) * b1[j]
              + (t[j + 3] - xv) / (t[j + 3] - t[j + 1]) * b1[j + 1];
#pragma unroll
    for (int j = 0; j < 8; ++j)
        b3[j] = (xv - t[j]) / (t[j + 3] - t[j]) * b2[j]
              + (t[j + 4] - xv) / (t[j + 4] - t[j + 1]) * b2[j + 1];

    float silu = xv / (1.f + __expf(-xv));

    _Float16* row = Apack + (size_t)b * K_TOTAL;
    row[i] = (_Float16)silu;

    v8h coef;
#pragma unroll
    for (int c = 0; c < NCOEF; ++c) coef[c] = (_Float16)b3[c];
    *(v8h*)(row + IN_F + (size_t)i * NCOEF) = coef;   // 16B aligned
}

// ------------------------------- weight packing ----------------------------
__global__ __launch_bounds__(256) void kan_pack_w(
    const float* __restrict__ bw, const float* __restrict__ sw,
    const float* __restrict__ scaler, _Float16* __restrict__ Wpack)
{
    int idx = blockIdx.x * 256 + threadIdx.x;        // one thread per (o, i)
    int o = idx >> 10;
    int i = idx & (IN_F - 1);
    if (o >= OUT_F) return;

    _Float16* row = Wpack + (size_t)o * K_TOTAL;
    row[i] = (_Float16)bw[(size_t)o * IN_F + i];

    float sc = scaler[(size_t)o * IN_F + i];
    const float* swp = sw + ((size_t)o * IN_F + i) * NCOEF;
    v8h coef;
#pragma unroll
    for (int c = 0; c < NCOEF; ++c) coef[c] = (_Float16)(swp[c] * sc);
    *(v8h*)(row + IN_F + (size_t)i * NCOEF) = coef;
}

// ------------------------- TDM tile descriptor + load ----------------------
// 2-D tile: 128 rows x 128B (16 units of 8B), row stride 2304 units in DRAM.
// pad_enable: after every 128B (pad_interval=4) insert 16B (pad_amount=3)
// -> LDS row stride 144B == LDK halves.
__device__ __forceinline__ void tdm_load_tile(
    const _Float16* gbase, int rowBase, int k0, int nRows,
    unsigned lds_byte_addr)
{
    unsigned long long ga =
        (unsigned long long)(size_t)(gbase + (size_t)rowBase * K_TOTAL + k0);

    v4u g0;
    g0[0] = 1u;                                           // count=1, user D#
    g0[1] = lds_byte_addr;                                // lds_addr
    g0[2] = (unsigned)(ga & 0xffffffffu);                 // global_addr lo
    g0[3] = (unsigned)((ga >> 32) & 0x1ffffffu)           // global_addr hi
          | (2u << 30);                                   // type = 2 (image)

    v8i g1;
    g1[0] = (3 << 16)      // data_size = 8B units
          | (1 << 20)      // pad_enable
          | (4 << 22)      // pad_interval: every 128B
          | (3 << 25);     // pad_amount: 16B
    g1[1] = (int)((2304u & 0xffffu) << 16);               // tensor_dim0 lo16
    g1[2] = (int)((2304u >> 16)                           // tensor_dim0 hi16
          | ((unsigned)(nRows & 0xffff) << 16));          // tensor_dim1 lo16
    g1[3] = (int)(((unsigned)nRows >> 16)                 // tensor_dim1 hi16
          | (16u << 16));                                 // tile_dim0 = 16 units
    g1[4] = 128;                                          // tile_dim1 = 128 rows
    g1[5] = 2304;                                         // tensor_dim0_stride lo
    g1[6] = 0;                                            // stride hi / dim1_stride lo
    g1[7] = 0;

    v4i gz4 = {0, 0, 0, 0};                               // 2-D: groups 2/3 unused
    v8i gz8 = {0, 0, 0, 0, 0, 0, 0, 0};
    __builtin_amdgcn_tensor_load_to_lds(g0, g1, gz4, gz4, gz8, 0);
}

// ------------------------------ WMMA GEMM ----------------------------------
// Block: 128 thr = 4 waves (2x2), tile 128(M) x 128(N), K stepped by 64.
// Wave tile 64x64 = 4x4 fragments -> 16 v_wmma per K32.
__device__ __forceinline__ v16h lds_frag(const _Float16* s, int row, int khalf)
{
    // ISA 7.12.2 16-bit layout: lane L holds matrix row (L&15),
    // halves 0..7 at K = (L>>4)*8, halves 8..15 at K = 16 + (L>>4)*8.
    const _Float16* p = s + row * LDK + khalf;
    v8h lo = *(const v8h*)(p);
    v8h hi = *(const v8h*)(p + 16);
    v16h f;
#pragma unroll
    for (int j = 0; j < 8; ++j) { f[j] = lo[j]; f[j + 8] = hi[j]; }
    return f;
}

__global__ __launch_bounds__(128) void kan_gemm(
    const _Float16* __restrict__ A, const _Float16* __restrict__ W,
    float* __restrict__ out)
{
    __shared__ _Float16 As[2][BM * LDK];   // 2 x 18KB
    __shared__ _Float16 Bs[2][BN * LDK];   // 2 x 18KB   (72KB total)

    const int tid   = threadIdx.x;
    const int lane  = tid & 31;
    const int wave  = tid >> 5;                  // 0..3
    const int waveM = wave >> 1;                 // 0..1
    const int waveN = wave & 1;                  // 0..1
    const int mBlock = blockIdx.y * BM;
    const int nBlock = blockIdx.x * BN;

    const int r  = lane & 15;                    // fragment row
    const int kh = (lane >> 4) * 8;              // K sub-offset (0 or 8)
    const int mOff = waveM * 64;                 // wave tile inside block
    const int nOff = waveN * 64;

    v8f acc[4][4] = {};

    // prologue: wave 0 DMAs stage 0 via the Tensor Data Mover
    if (wave == 0) {
        tdm_load_tile(A, mBlock, 0, BATCH, (unsigned)(size_t)&As[0][0]);
        tdm_load_tile(W, nBlock, 0, OUT_F, (unsigned)(size_t)&Bs[0][0]);
    }

    for (int k0 = 0; k0 < K_TOTAL; k0 += BK) {
        const int cur = (k0 / BK) & 1;
        const int nxt = cur ^ 1;
        const bool more = (k0 + BK) < K_TOTAL;

        if (wave == 0) {
            if (more) {                            // prefetch next stage
                tdm_load_tile(A, mBlock, k0 + BK, BATCH,
                              (unsigned)(size_t)&As[nxt][0]);
                tdm_load_tile(W, nBlock, k0 + BK, OUT_F,
                              (unsigned)(size_t)&Bs[nxt][0]);
                // 4 in flight; tensor ops retire in order -> <=2 means cur done
                __builtin_amdgcn_s_wait_tensorcnt(2);
            } else {
                __builtin_amdgcn_s_wait_tensorcnt(0);
            }
        }
        __syncthreads();                           // cur tile visible to all

#pragma unroll
        for (int kk = 0; kk < BK; kk += 32) {
            v16h af[4], bf[4];
#pragma unroll
            for (int mi = 0; mi < 4; ++mi)
                af[mi] = lds_frag(As[cur], mOff + mi * 16 + r, kk + kh);
#pragma unroll
            for (int ni = 0; ni < 4; ++ni)
                bf[ni] = lds_frag(Bs[cur], nOff + ni * 16 + r, kk + kh);

#pragma unroll
            for (int mi = 0; mi < 4; ++mi)
#pragma unroll
                for (int ni = 0; ni < 4; ++ni)
                    acc[mi][ni] = __builtin_amdgcn_wmma_f32_16x16x32_f16(
                        false, af[mi], false, bf[ni],
                        (short)0, acc[mi][ni], false, false);
        }
        __syncthreads();                           // done reading cur buffer
    }

    // C/D layout: VGPR v holds (M = v + 8*(lane>>4), N = lane&15)
    const int mo = (lane >> 4) * 8;
    const int co = lane & 15;
#pragma unroll
    for (int mi = 0; mi < 4; ++mi)
#pragma unroll
        for (int ni = 0; ni < 4; ++ni)
#pragma unroll
            for (int v = 0; v < 8; ++v)
                out[(size_t)(mBlock + mOff + mi * 16 + mo + v) * OUT_F
                    + nBlock + nOff + ni * 16 + co] = acc[mi][ni][v];
}

// ------------------------------- launcher ----------------------------------
extern "C" void kernel_launch(void* const* d_in, const int* in_sizes, int n_in,
                              void* d_out, int out_size, void* d_ws, size_t ws_size,
                              hipStream_t stream)
{
    const float* x      = (const float*)d_in[0];   // (4096, 1024)
    const float* grid   = (const float*)d_in[1];   // (1024, 12)
    const float* bw     = (const float*)d_in[2];   // (1024, 1024)
    const float* sw     = (const float*)d_in[3];   // (1024, 1024, 8)
    const float* scaler = (const float*)d_in[4];   // (1024, 1024)

    _Float16* Apack = (_Float16*)d_ws;
    _Float16* Wpack = (_Float16*)((char*)d_ws + A_BYTES);

    kan_pack_act<<<(BATCH * IN_F) / 256, 256, 0, stream>>>(x, grid, Apack);
    kan_pack_w<<<(OUT_F * IN_F) / 256, 256, 0, stream>>>(bw, sw, scaler, Wpack);

    dim3 grd(OUT_F / BN, BATCH / BM);              // (8, 32)
    kan_gemm<<<grd, 128, 0, stream>>>(Apack, Wpack, (float*)d_out);
}
